// VectorQuantizerEMA_15771119911011
// MI455X (gfx1250) — matrix operand order repaired and checked by reading, not compiled, not run on hardware
//
#include <hip/hip_runtime.h>

typedef __attribute__((ext_vector_type(16))) _Float16 v16h;
typedef __attribute__((ext_vector_type(8)))  _Float16 v8h;
typedef __attribute__((ext_vector_type(8)))  float    v8f;
typedef __attribute__((ext_vector_type(4)))  int      v4i;

// Problem constants
#define K_CODES   2048
#define D_DIM     64
#define N_ROWS    131072            // 32*4096
#define N_ELEMS   8388608           // N_ROWS * D_DIM
#define EMA_DECAY 0.99f
#define ONE_M_DECAY 0.01f
#define EPSILON   1e-5f
#define COMMIT_COST 0.25f

// argmin kernel tiling
#define CODE_TILE 256               // codes resident in LDS per chunk
#define N_CHUNKS  (K_CODES / CODE_TILE)
#define LDS_ROW   72                // padded row stride in halves (64 + 8)
#define S_TILES   4                 // 16-row WMMA tiles per wave (64 rows/wave)
#define ROWS_PER_BLOCK (8 * 16 * S_TILES)   // 8 waves * 16 rows * S_TILES = 512
// double-buffered: 2 chunks + 2 norm arrays
#define ARGMIN_LDS_BYTES (2*CODE_TILE*LDS_ROW*2 + 2*CODE_TILE*4)   // 75776 B

#if __has_builtin(__builtin_amdgcn_global_load_async_to_lds_b128) && \
    __has_builtin(__builtin_amdgcn_global_load_async_to_lds_b32)
#define HAS_ASYNC_LDS 1
#else
#define HAS_ASYNC_LDS 0
#endif

#if HAS_ASYNC_LDS
// Signatures per hipcc diagnostics: arg0 = AS(1) data-typed pointer (non-const),
// arg1 = AS(3) LDS pointer, then imm offset + imm cpol.
static __device__ __forceinline__ void async_cp_b128(const void* g, void* l) {
  __builtin_amdgcn_global_load_async_to_lds_b128(
      (__attribute__((address_space(1))) v4i*)g,
      (__attribute__((address_space(3))) v4i*)l, 0, 0);
}
static __device__ __forceinline__ void async_cp_b32(const void* g, void* l) {
  __builtin_amdgcn_global_load_async_to_lds_b32(
      (__attribute__((address_space(1))) int*)g,
      (__attribute__((address_space(3))) int*)l, 0, 0);
}
#endif

static __device__ __forceinline__ void wait_async0() {
#if __has_builtin(__builtin_amdgcn_s_wait_asynccnt)
  __builtin_amdgcn_s_wait_asynccnt(0);
#else
  asm volatile("s_wait_asynccnt 0x0" ::: "memory");
#endif
}

// ---------------------------------------------------------------------------
// Kernel 1: prep — e-norms, f16 codebook, zero accumulators (every call)
// ---------------------------------------------------------------------------
__global__ __launch_bounds__(256) void vq_prep(const float* __restrict__ emb,
                                               _Float16* __restrict__ emb16,
                                               float* __restrict__ enorm,
                                               float* __restrict__ count_ws,
                                               float* __restrict__ sum_ws,
                                               float* __restrict__ loss_ws) {
  const int c = blockIdx.x * 256 + threadIdx.x;
  if (c == 0) loss_ws[0] = 0.0f;
  if (c >= K_CODES) return;
  float acc = 0.0f;
  const float* row = emb + (size_t)c * D_DIM;
  _Float16* orow = emb16 + (size_t)c * D_DIM;
  float* zrow = sum_ws + (size_t)c * D_DIM;
  const float4 zero4 = make_float4(0.f, 0.f, 0.f, 0.f);
#pragma unroll
  for (int q = 0; q < 16; ++q) {
    float4 v = *(const float4*)(row + q * 4);
    acc += v.x*v.x + v.y*v.y + v.z*v.z + v.w*v.w;
    orow[q*4 + 0] = (_Float16)v.x;
    orow[q*4 + 1] = (_Float16)v.y;
    orow[q*4 + 2] = (_Float16)v.z;
    orow[q*4 + 3] = (_Float16)v.w;
    *(float4*)(zrow + q * 4) = zero4;
  }
  enorm[c] = -0.5f * acc;     // pre-negate/halve: WMMA C-init gives score = z.e - |e|^2/2
  count_ws[c] = 0.0f;
}

// ---------------------------------------------------------------------------
// Chunk loader: CODE_TILE codebook rows (f16) + (-0.5*norm) into an LDS buffer.
// One code row (128 B) per thread.
// ---------------------------------------------------------------------------
static __device__ __forceinline__ void load_chunk(const _Float16* __restrict__ emb16,
                                                  const float* __restrict__ enorm,
                                                  int chunkBase, int t,
                                                  _Float16* sembBuf, float* snormBuf) {
  const uint4* src = (const uint4*)(emb16 + (size_t)(chunkBase + t) * D_DIM);
  uint4* dst = (uint4*)(sembBuf + t * LDS_ROW);
#if HAS_ASYNC_LDS
#pragma unroll
  for (int q = 0; q < 8; ++q) async_cp_b128(src + q, dst + q);
  async_cp_b32(enorm + chunkBase + t, snormBuf + t);
#else
  uint4 u[8];
#pragma unroll
  for (int q = 0; q < 8; ++q) u[q] = src[q];          // pipelined loads
#pragma unroll
  for (int q = 0; q < 8; ++q) dst[q] = u[q];
  snormBuf[t] = enorm[chunkBase + t];
#endif
}

// ---------------------------------------------------------------------------
// Kernel 2: WMMA argmin over 2048 codes
//   block = 256 threads = 8 waves; wave handles 64 rows (4 WMMA row-tiles)
//   grid  = N_ROWS / 512 = 256 blocks; codebook double-buffered through LDS
//   score(m,c) = z.e - |e|^2/2  (C-init = -|e|^2/2)  -> argmax == dist argmin
// ---------------------------------------------------------------------------
__global__ __launch_bounds__(256) void vq_argmin(const float* __restrict__ z,
                                                 const _Float16* __restrict__ emb16,
                                                 const float* __restrict__ enorm,
                                                 int* __restrict__ idx_ws,
                                                 float* __restrict__ idx_out) {
  extern __shared__ char smem[];
  _Float16* semb0 = (_Float16*)smem;
  _Float16* semb1 = semb0 + CODE_TILE * LDS_ROW;
  float* snorm0 = (float*)(smem + 2 * CODE_TILE * LDS_ROW * sizeof(_Float16));
  float* snorm1 = snorm0 + CODE_TILE;

  const int t    = threadIdx.x;
  const int lane = t & 31;
  const int wave = t >> 5;
  const int n    = lane & 15;       // column (code) within 16-tile / row within A tile
  const int hi   = lane >> 4;       // upper/lower 16-lane half
  const int rowBase = blockIdx.x * ROWS_PER_BLOCK + wave * (16 * S_TILES);

  // kick off chunk 0 while we build A fragments
  load_chunk(emb16, enorm, 0, t, semb0, snorm0);

  // ---- A fragments: persist across the whole K sweep ----------------------
  // 16-bit A 16x32 layout (ISA 7.12.2): lane half selects +8 k-offset,
  // halves 0..7 -> k0..k0+7, halves 8..15 -> k0+16..k0+23.
  v16h a[S_TILES][2];
#pragma unroll
  for (int s = 0; s < S_TILES; ++s) {
    const float* zrow = z + (size_t)(rowBase + s * 16 + n) * D_DIM;
#pragma unroll
    for (int kb = 0; kb < 2; ++kb) {
      const int k0 = kb * 32 + hi * 8;
      float4 f0 = *(const float4*)(zrow + k0);
      float4 f1 = *(const float4*)(zrow + k0 + 4);
      float4 f2 = *(const float4*)(zrow + k0 + 16);
      float4 f3 = *(const float4*)(zrow + k0 + 20);
      v16h av;
      av[0]=(_Float16)f0.x;  av[1]=(_Float16)f0.y;  av[2]=(_Float16)f0.z;  av[3]=(_Float16)f0.w;
      av[4]=(_Float16)f1.x;  av[5]=(_Float16)f1.y;  av[6]=(_Float16)f1.z;  av[7]=(_Float16)f1.w;
      av[8]=(_Float16)f2.x;  av[9]=(_Float16)f2.y;  av[10]=(_Float16)f2.z; av[11]=(_Float16)f2.w;
      av[12]=(_Float16)f3.x; av[13]=(_Float16)f3.y; av[14]=(_Float16)f3.z; av[15]=(_Float16)f3.w;
      a[s][kb] = av;
    }
  }

  float maxv[S_TILES][8];
  int   maxi[S_TILES][8];
#pragma unroll
  for (int s = 0; s < S_TILES; ++s)
#pragma unroll
    for (int j = 0; j < 8; ++j) { maxv[s][j] = -3.402823466e38f; maxi[s][j] = 0; }

  // ---- sweep codebook: double-buffered LDS chunks --------------------------
  for (int chunk = 0; chunk < N_CHUNKS; ++chunk) {
    const int cur = chunk & 1;
    _Float16* sembCur = cur ? semb1 : semb0;
    float*    snormCur = cur ? snorm1 : snorm0;

    wait_async0();        // my loads for this chunk complete
    __syncthreads();      // everyone's loads visible; prior readers of other buf done

    if (chunk + 1 < N_CHUNKS)   // prefetch next chunk into the other buffer
      load_chunk(emb16, enorm, (chunk + 1) * CODE_TILE, t,
                 cur ? semb0 : semb1, cur ? snorm0 : snorm1);

    const int chunkBase = chunk * CODE_TILE;
    for (int ct = 0; ct < CODE_TILE / 16; ++ct) {
      // B 32x16 16-bit layout (matches A): lane half selects +8 k-offset,
      // halves 0..7 -> k0..k0+7, halves 8..15 -> k0+16..k0+23.
      const _Float16* srow = sembCur + (ct * 16 + n) * LDS_ROW + hi * 8;
      v16h b0, b1;
      *(v8h*)&b0       = *(const v8h*)(srow);
      *((v8h*)&b0 + 1) = *(const v8h*)(srow + 16);
      *(v8h*)&b1       = *(const v8h*)(srow + 32);
      *((v8h*)&b1 + 1) = *(const v8h*)(srow + 48);

      const float nrm  = snormCur[ct * 16 + n];   // = -0.5*|e|^2 for this lane's column
      const int   code = chunkBase + ct * 16 + n;
#pragma unroll
      for (int s = 0; s < S_TILES; ++s) {
        v8f c;
#pragma unroll
        for (int j = 0; j < 8; ++j) c[j] = nrm;   // C-init: -|e|^2/2
        c = __builtin_amdgcn_wmma_f32_16x16x32_f16(false, a[s][0], false, b0,
                                                   (short)0, c, false, false);
        c = __builtin_amdgcn_wmma_f32_16x16x32_f16(false, a[s][1], false, b1,
                                                   (short)0, c, false, false);
#pragma unroll
        for (int j = 0; j < 8; ++j) {
          if (c[j] > maxv[s][j]) { maxv[s][j] = c[j]; maxi[s][j] = code; } // first-index tie-break
        }
      }
    }
  }

  // ---- reduce over the 16 lanes of each half (C layout: lane = column N) --
#pragma unroll
  for (int off = 8; off >= 1; off >>= 1) {
#pragma unroll
    for (int s = 0; s < S_TILES; ++s)
#pragma unroll
      for (int j = 0; j < 8; ++j) {
        const float ov = __shfl_xor(maxv[s][j], off, 32);
        const int   oi = __shfl_xor(maxi[s][j], off, 32);
        if (ov > maxv[s][j] || (ov == maxv[s][j] && oi < maxi[s][j])) {
          maxv[s][j] = ov; maxi[s][j] = oi;
        }
      }
  }

  if (n == 0) {   // lanes 0 and 16: C rows M = j + 8*hi
#pragma unroll
    for (int s = 0; s < S_TILES; ++s)
#pragma unroll
      for (int j = 0; j < 8; ++j) {
        const int row = rowBase + s * 16 + j + hi * 8;
        idx_ws[row]  = maxi[s][j];
        idx_out[row] = (float)maxi[s][j];
      }
  }
}

// ---------------------------------------------------------------------------
// Kernel 3: gather z_q, straight-through output, scatter EMA stats, loss
// ---------------------------------------------------------------------------
__global__ __launch_bounds__(256) void vq_scatter(const float* __restrict__ z,
                                                  const float* __restrict__ emb,
                                                  const int* __restrict__ idx_ws,
                                                  float* __restrict__ zq_out,
                                                  float* __restrict__ sum_ws,
                                                  float* __restrict__ count_ws,
                                                  float* __restrict__ loss_ws) {
  __shared__ float swred[8];
  const int e   = blockIdx.x * 256 + threadIdx.x;     // < N_ELEMS
  const int row = e >> 6;
  const int d   = e & 63;
  const int c   = idx_ws[row];
  const float zv = z[e];
  const float q  = emb[c * D_DIM + d];
  zq_out[e] = zv + (q - zv);                          // matches z + sg(z_q - z) bit pattern
  const float diff = zv - q;
  atomicAdd(&sum_ws[c * D_DIM + d], zv);
  if (d == 0) atomicAdd(&count_ws[c], 1.0f);

  // wave32 reduce, then tiny LDS combine, one atomic per block
  float v = diff * diff;
#pragma unroll
  for (int off = 16; off >= 1; off >>= 1) v += __shfl_xor(v, off, 32);
  if ((threadIdx.x & 31) == 0) swred[threadIdx.x >> 5] = v;
  __syncthreads();
  if (threadIdx.x == 0) {
    float s = 0.0f;
#pragma unroll
    for (int w = 0; w < 8; ++w) s += swred[w];
    atomicAdd(loss_ws, s);
  }
}

// ---------------------------------------------------------------------------
// Kernel 4: EMA finalize, smoothing, loss + perplexity scalars (one block)
// ---------------------------------------------------------------------------
__global__ __launch_bounds__(1024) void vq_finalize(const float* __restrict__ ema_cs,
                                                    const float* __restrict__ ema_sum,
                                                    const float* __restrict__ count_ws,
                                                    const float* __restrict__ sum_ws,
                                                    const float* __restrict__ loss_ws,
                                                    float* __restrict__ out_loss,
                                                    float* __restrict__ out_perp,
                                                    float* __restrict__ out_new_emb,
                                                    float* __restrict__ out_new_cs,
                                                    float* __restrict__ out_new_sum) {
  __shared__ float sn[1024];
  __shared__ float se[1024];
  const int t = threadIdx.x;
  float nacc = 0.0f, eacc = 0.0f;
  for (int c = t; c < K_CODES; c += 1024) {
    const float cnt = count_ws[c];
    const float ncs = EMA_DECAY * ema_cs[c] + ONE_M_DECAY * cnt;
    out_new_cs[c] = ncs;
    nacc += ncs;
    const float p = cnt * (1.0f / (float)N_ROWS);
    eacc += p * logf(p + 1e-10f);
  }
  sn[t] = nacc; se[t] = eacc;
  __syncthreads();
  for (int s = 512; s >= 1; s >>= 1) {
    if (t < s) { sn[t] += sn[t + s]; se[t] += se[t + s]; }
    __syncthreads();
  }
  const float nsum = sn[0];
  if (t == 0) {
    out_loss[0] = COMMIT_COST * loss_ws[0] * (1.0f / (float)N_ELEMS);
    out_perp[0] = expf(-se[0]);
  }
  const float denom = nsum + (float)K_CODES * EPSILON;
  for (int e = t; e < K_CODES * D_DIM; e += 1024) {
    const int c = e >> 6;
    const float ncs = out_new_cs[c];
    const float css = (ncs + EPSILON) / denom * nsum;
    const float nes = EMA_DECAY * ema_sum[e] + ONE_M_DECAY * sum_ws[e];
    out_new_sum[e] = nes;
    out_new_emb[e] = nes / css;
  }
}

// ---------------------------------------------------------------------------
extern "C" void kernel_launch(void* const* d_in, const int* in_sizes, int n_in,
                              void* d_out, int out_size, void* d_ws, size_t ws_size,
                              hipStream_t stream) {
  (void)in_sizes; (void)n_in; (void)out_size; (void)ws_size;

  const float* z       = (const float*)d_in[0];
  const float* emb     = (const float*)d_in[1];
  const float* ema_cs  = (const float*)d_in[2];
  const float* ema_sum = (const float*)d_in[3];

  float* out = (float*)d_out;
  float* out_zq      = out;                       // 8388608
  float* out_idx     = out + 8388608;             // 131072
  float* out_loss    = out + 8519680;             // 1
  float* out_perp    = out + 8519681;             // 1
  float* out_new_emb = out + 8519682;             // 131072
  float* out_new_cs  = out + 8650754;             // 2048
  float* out_new_sum = out + 8652802;             // 131072

  char* ws = (char*)d_ws;
  _Float16* emb16  = (_Float16*)(ws);             // 262144 B
  float* enorm     = (float*)(ws + 262144);       // 8192  B  (stores -0.5*|e|^2)
  int*   idx_ws    = (int*)  (ws + 270336);       // 524288 B
  float* count_ws  = (float*)(ws + 794624);       // 8192  B
  float* sum_ws    = (float*)(ws + 802816);       // 524288 B
  float* loss_ws   = (float*)(ws + 1327104);      // 4 B

  vq_prep<<<K_CODES / 256, 256, 0, stream>>>(emb, emb16, enorm, count_ws, sum_ws, loss_ws);

  vq_argmin<<<N_ROWS / ROWS_PER_BLOCK, 256, ARGMIN_LDS_BYTES, stream>>>(z, emb16, enorm,
                                                                        idx_ws, out_idx);

  vq_scatter<<<N_ELEMS / 256, 256, 0, stream>>>(z, emb, idx_ws, out_zq, sum_ws, count_ws, loss_ws);

  vq_finalize<<<1, 1024, 0, stream>>>(ema_cs, ema_sum, count_ws, sum_ws, loss_ws,
                                      out_loss, out_perp, out_new_emb, out_new_cs, out_new_sum);
}
